// GemmaAttention_57354993271004
// MI455X (gfx1250) — compile-verified, matrix operand order checked by reading, not run on hardware
//
#include <hip/hip_runtime.h>

typedef __attribute__((ext_vector_type(16))) __bf16 v16bf;
typedef __attribute__((ext_vector_type(8)))  __bf16 v8bf;
typedef __attribute__((ext_vector_type(8)))  float  v8f;
typedef __attribute__((ext_vector_type(4)))  unsigned int u32x4;
typedef __attribute__((ext_vector_type(8)))  int      i32x8;
typedef __attribute__((ext_vector_type(4)))  int      i32x4;

static constexpr int B_   = 2;
static constexpr int S_   = 2048;
static constexpr int DIM_ = 2048;
static constexpr int NH_  = 8;
static constexpr int HD_  = 256;
static constexpr int M_   = B_ * S_;   // 4096 token rows

#if defined(__HIP_DEVICE_COMPILE__) && __has_builtin(__builtin_amdgcn_tensor_load_to_lds) && __has_builtin(__builtin_amdgcn_s_wait_tensorcnt)
#define USE_TDM 1
#else
#define USE_TDM 0
#endif

__device__ __forceinline__ unsigned short f32_to_bf16(float f) {
  unsigned int u = __float_as_uint(f);
  u += 0x7FFFu + ((u >> 16) & 1u);     // round-to-nearest-even
  return (unsigned short)(u >> 16);
}

// Fragment loader for 16-bit A/B operands of V_WMMA_*_16X16X32:
// per-lane data is two contiguous 8-element (16B) groups at p and p+16,
// where p already includes the per-lane (row*K + 8*half) offset.
__device__ __forceinline__ v16bf ld_frag(const unsigned short* p) {
  union { v16bf v; v8bf h[2]; } u;
  u.h[0] = *(const v8bf*)(p);
  u.h[1] = *(const v8bf*)(p + 16);
  return u.v;
}

__device__ __forceinline__ v8f wmma_bf16(v16bf a, v16bf b, v8f c) {
  return __builtin_amdgcn_wmma_f32_16x16x32_bf16(false, a, false, b, (short)0, c,
                                                 false, false);
}

#if USE_TDM
// TDM descriptor (D#) per CDNA5 ISA ch.8: 2D tile, 32 rows x 256 bf16 (= one
// 32-key K chunk, row stride 256 elements), contiguous into LDS.
__device__ __forceinline__ void tdm_load_k32(unsigned lds_byte,
                                             unsigned long long gaddr) {
  u32x4 g0 = { 1u,                                   // count=1, user mode
               lds_byte,                             // lds_addr
               (unsigned)gaddr,                      // global_addr[31:0]
               (unsigned)((gaddr >> 32) & 0x01FFFFFFu) | (2u << 30) }; // type=2
  i32x8 g1 = { 0x00010000,   // data_size=2B, mask=0
               0x01000000,   // tensor_dim0 = 256 (low16 << 16)
               0x00200000,   // tensor_dim1 = 32  (low16 << 16)
               0x01000000,   // tile_dim0   = 256 (<<16)
               32,           // tile_dim1   = 32, tile_dim2 = 0
               256,          // tensor_dim0_stride = 256 elements
               0, 0 };
  i32x4 z4 = { 0, 0, 0, 0 };
#if __clang_major__ >= 23
  i32x8 z8 = { 0, 0, 0, 0, 0, 0, 0, 0 };
  __builtin_amdgcn_tensor_load_to_lds(g0, g1, z4, z4, z8, 0);
#else
  __builtin_amdgcn_tensor_load_to_lds(g0, g1, z4, z4, 0);
#endif
}
#endif

// ---------------- RMSNorm: x[row,:] -> bf16 xn with (1+gains) scale ----------
__global__ void rmsnorm_bf16_kernel(const float* __restrict__ x,
                                    const float* __restrict__ gains,
                                    unsigned short* __restrict__ xn) {
  __shared__ float red[256];
  const int row = blockIdx.x;
  const int tid = threadIdx.x;
  const float* xr = x + (size_t)row * DIM_;
  float ss = 0.f;
  for (int c = tid; c < DIM_; c += 256) { float v = xr[c]; ss += v * v; }
  red[tid] = ss;
  __syncthreads();
  for (int o = 128; o > 0; o >>= 1) {
    if (tid < o) red[tid] += red[tid + o];
    __syncthreads();
  }
  const float scale = rsqrtf(red[0] * (1.0f / (float)DIM_) + 1e-6f);
  unsigned short* orow = xn + (size_t)row * DIM_;
  for (int c = tid; c < DIM_; c += 256)
    orow[c] = f32_to_bf16(xr[c] * scale * (1.f + gains[c]));
}

// ---------------- fp32 -> bf16 weight conversion ----------------------------
__global__ void cvt_bf16_kernel(const float* __restrict__ in,
                                unsigned short* __restrict__ outp, int n) {
  int i = blockIdx.x * 256 + threadIdx.x;
  if (i < n) outp[i] = f32_to_bf16(in[i]);
}

// ---------------- Generic bf16 WMMA GEMM: C[M,N] = A[M,K] * Bw[N,K]^T -------
// Block: 256 threads = 8 waves arranged 2(M) x 4(N). Each wave owns a 32x32
// output tile (4 independent accumulators -> 4 independent WMMA chains).
// Block tile: 64(M) x 128(N).
__global__ void gemm_bf16_wmma_kernel(const unsigned short* __restrict__ A,
                                      const unsigned short* __restrict__ Bw,
                                      float* __restrict__ Cf,
                                      unsigned short* __restrict__ Cbf,
                                      const float* __restrict__ resid,
                                      int M, int N, int K) {
  const int w    = threadIdx.x >> 5;
  const int lane = threadIdx.x & 31;
  const int wm   = w >> 2, wn = w & 3;
  const int row0 = blockIdx.y * 64 + wm * 32;
  const int col0 = blockIdx.x * 128 + wn * 32;
  const int mi   = lane & 15;
  const int half = lane >> 4;
  const unsigned short* arow0 = A  + (size_t)(row0 + mi) * K + 8 * half;
  const unsigned short* arow1 = arow0 + (size_t)16 * K;
  const unsigned short* brow0 = Bw + (size_t)(col0 + mi) * K + 8 * half;
  const unsigned short* brow1 = brow0 + (size_t)16 * K;
  v8f acc00 = {0.f,0.f,0.f,0.f,0.f,0.f,0.f,0.f};
  v8f acc01 = acc00, acc10 = acc00, acc11 = acc00;
#pragma unroll 2
  for (int k0 = 0; k0 < K; k0 += 32) {
    const v16bf a0 = ld_frag(arow0 + k0);
    const v16bf a1 = ld_frag(arow1 + k0);
    const v16bf b0 = ld_frag(brow0 + k0);
    const v16bf b1 = ld_frag(brow1 + k0);
    acc00 = wmma_bf16(a0, b0, acc00);
    acc01 = wmma_bf16(a0, b1, acc01);
    acc10 = wmma_bf16(a1, b0, acc10);
    acc11 = wmma_bf16(a1, b1, acc11);
  }
  const int n = lane & 15;
  v8f accs[2][2] = {{acc00, acc01}, {acc10, acc11}};
  for (int tm = 0; tm < 2; ++tm)
    for (int tn = 0; tn < 2; ++tn) {
      v8f acc = accs[tm][tn];
#pragma unroll
      for (int r = 0; r < 8; ++r) {
        const int mm = tm * 16 + r + 8 * half;
        const size_t o = (size_t)(row0 + mm) * N + col0 + tn * 16 + n;
        float vv = acc[r];
        if (resid) vv += resid[o];
        if (Cf)  Cf[o]  = vv;
        if (Cbf) Cbf[o] = f32_to_bf16(vv);
      }
    }
}

// ---------------- RoPE + layout packing -------------------------------------
// out[t]     = x[t]*cos - x[t+128]*sin
// out[t+128] = x[t]*sin + x[t+128]*cos   (theta shared per interleaved pair)
__global__ void rope_pack_kernel(const float* __restrict__ q,
                                 const float* __restrict__ k,
                                 const float* __restrict__ v,
                                 const float* __restrict__ sinv,
                                 const float* __restrict__ cosv,
                                 unsigned short* __restrict__ q_rot,  // [B,NH,S,HD]
                                 unsigned short* __restrict__ k_rot,  // [B,S,HD]
                                 unsigned short* __restrict__ vT) {   // [B,HD,S]
  const int s = blockIdx.x, b = blockIdx.y;
  const int t = threadIdx.x;  // 0..127
  const float sn = sinv[(size_t)s * HD_ + 2 * t];
  const float cs = cosv[(size_t)s * HD_ + 2 * t];
  const size_t qrow = ((size_t)b * S_ + s) * DIM_;
  for (int h = 0; h < NH_; ++h) {
    const float x0 = q[qrow + h * HD_ + t];
    const float x1 = q[qrow + h * HD_ + t + 128];
    const size_t ob = ((size_t)(b * NH_ + h) * S_ + s) * HD_;
    q_rot[ob + t]       = f32_to_bf16(x0 * cs - x1 * sn);
    q_rot[ob + t + 128] = f32_to_bf16(x0 * sn + x1 * cs);
  }
  const size_t krow = ((size_t)b * S_ + s) * HD_;
  const float k0v = k[krow + t], k1v = k[krow + t + 128];
  k_rot[krow + t]       = f32_to_bf16(k0v * cs - k1v * sn);
  k_rot[krow + t + 128] = f32_to_bf16(k0v * sn + k1v * cs);
  const float v0 = v[krow + t], v1 = v[krow + t + 128];
  vT[((size_t)b * HD_ + t)       * S_ + s] = f32_to_bf16(v0);
  vT[((size_t)b * HD_ + t + 128) * S_ + s] = f32_to_bf16(v1);
}

// ---------------- Flash attention: one wave per 16-query tile ---------------
// K chunks (32 keys x 256) are TDM-prefetched into double-buffered LDS,
// synchronized with s_wait_tensorcnt (in-order), when TDM builtins exist.
__global__ void flash_attn_wmma_kernel(const unsigned short* __restrict__ q_rot,
                                       const unsigned short* __restrict__ k_rot,
                                       const unsigned short* __restrict__ vT,
                                       unsigned short* __restrict__ attn_bf) {
  __shared__ __align__(16) float          s_tile[16][32];
  __shared__ __align__(16) unsigned short p_tile[16][32];
  __shared__ float m_s[16], l_s[16], fac[16];
#if USE_TDM
  __shared__ __align__(16) unsigned short kbuf[2][32 * HD_];
#endif

  const int tile = blockIdx.x;
  const int h    = blockIdx.y;
  const int b    = blockIdx.z;
  const int lane = threadIdx.x;
  const int mi   = lane & 15;
  const int half = lane >> 4;
  const int row0 = tile * 16;
  const float scale = 0.0625f;  // 1/sqrt(HD)

  const unsigned short* qbase = q_rot + ((size_t)(b * NH_ + h) * S_) * HD_;
  const unsigned short* kbase = k_rot + ((size_t)b * S_) * HD_;
  const unsigned short* vbase = vT    + ((size_t)b * HD_) * S_;

  // Q fragments: 8 chunks of K=32 covering HD=256; kept in registers.
  v16bf qa[8];
  {
    const unsigned short* qrow = qbase + (size_t)(row0 + mi) * HD_ + 8 * half;
#pragma unroll
    for (int c = 0; c < 8; ++c) qa[c] = ld_frag(qrow + c * 32);
  }

  v8f O[16];
#pragma unroll
  for (int f = 0; f < 16; ++f)
    O[f] = (v8f){0.f, 0.f, 0.f, 0.f, 0.f, 0.f, 0.f, 0.f};

  if (lane < 16) { m_s[lane] = -1e30f; l_s[lane] = 0.f; }
  __syncthreads();

  const int nchunks = (row0 + 16 + 31) / 32;  // causal: skip fully-masked chunks
#if USE_TDM
  tdm_load_k32((unsigned)(uintptr_t)&kbuf[0][0],
               (unsigned long long)(uintptr_t)kbase);
#endif
  for (int kc = 0; kc < nchunks; ++kc) {
    const int key0 = kc * 32;
#if USE_TDM
    if (kc + 1 < nchunks) {
      tdm_load_k32((unsigned)(uintptr_t)&kbuf[(kc + 1) & 1][0],
                   (unsigned long long)(uintptr_t)
                       (kbase + (size_t)(key0 + 32) * HD_));
      __builtin_amdgcn_s_wait_tensorcnt(1);   // current chunk landed (in-order)
    } else {
      __builtin_amdgcn_s_wait_tensorcnt(0);
    }
    asm volatile("" ::: "memory");            // keep LDS reads behind the wait
    const unsigned short* kchunk = &kbuf[kc & 1][0];
    const unsigned short* krow0 = kchunk + (size_t)mi * HD_ + 8 * half;
    const unsigned short* krow1 = kchunk + (size_t)(16 + mi) * HD_ + 8 * half;
#else
    const unsigned short* krow0 = kbase + (size_t)(key0 + mi) * HD_ + 8 * half;
    const unsigned short* krow1 = kbase + (size_t)(key0 + 16 + mi) * HD_ + 8 * half;
#endif
    // scores for two 16-key sub-tiles, interleaved (2 independent chains)
    {
      v8f acc0 = {0.f,0.f,0.f,0.f,0.f,0.f,0.f,0.f};
      v8f acc1 = acc0;
#pragma unroll
      for (int c = 0; c < 8; ++c) {
        acc0 = wmma_bf16(qa[c], ld_frag(krow0 + c * 32), acc0);
        acc1 = wmma_bf16(qa[c], ld_frag(krow1 + c * 32), acc1);
      }
#pragma unroll
      for (int r = 0; r < 8; ++r) {
        s_tile[r + 8 * half][mi]      = acc0[r];
        s_tile[r + 8 * half][16 + mi] = acc1[r];
      }
    }
    __syncthreads();

    // online softmax; lane t owns query row t
    if (lane < 16) {
      const int qidx = row0 + lane;
      const float mx = m_s[lane];
      float tmax = -1e30f;
      for (int j = 0; j < 32; ++j) {
        float sv = s_tile[lane][j] * scale;
        if (key0 + j > qidx) sv = -1e30f;    // causal mask
        s_tile[lane][j] = sv;
        tmax = fmaxf(tmax, sv);
      }
      const float newm = fmaxf(mx, tmax);
      const float fct  = __expf(mx - newm);
      float rs = 0.f;
      for (int j = 0; j < 32; ++j) {
        const float e = __expf(s_tile[lane][j] - newm);
        rs += e;
        p_tile[lane][j] = f32_to_bf16(e);
      }
      m_s[lane] = newm;
      l_s[lane] = l_s[lane] * fct + rs;
      fac[lane] = fct;
    }
    __syncthreads();

    float fr[8];
#pragma unroll
    for (int r = 0; r < 8; ++r) fr[r] = fac[r + 8 * half];
    const v16bf pa = ld_frag(&p_tile[mi][8 * half]);  // P as A-fragment (K=32 keys)
#pragma unroll
    for (int f = 0; f < 16; ++f) {
      const unsigned short* vcol =
          vbase + (size_t)(f * 16 + mi) * S_ + key0 + 8 * half;
      const v16bf vb = ld_frag(vcol);
      v8f o = O[f];
#pragma unroll
      for (int r = 0; r < 8; ++r) o[r] *= fr[r];
      O[f] = wmma_bf16(pa, vb, o);
    }
    __syncthreads();
  }

  if (lane < 16) fac[lane] = 1.f / l_s[lane];
  __syncthreads();
  float inv[8];
#pragma unroll
  for (int r = 0; r < 8; ++r) inv[r] = fac[r + 8 * half];

  // write [b,s, h*HD + feat] as bf16 for the WO GEMM
  unsigned short* obase = attn_bf + ((size_t)(b * S_ + row0)) * DIM_ + h * HD_;
  const int n = lane & 15;
  for (int f = 0; f < 16; ++f)
#pragma unroll
    for (int r = 0; r < 8; ++r)
      obase[(size_t)(r + 8 * half) * DIM_ + f * 16 + n] =
          f32_to_bf16(O[f][r] * inv[r]);
}

// ---------------- host-side orchestration -----------------------------------
extern "C" void kernel_launch(void* const* d_in, const int* in_sizes, int n_in,
                              void* d_out, int out_size, void* d_ws, size_t ws_size,
                              hipStream_t stream) {
  (void)in_sizes; (void)n_in; (void)out_size; (void)ws_size;
  const float* x     = (const float*)d_in[0];
  // d_in[1] att_mask: causal, applied analytically in the flash kernel
  const float* sinv  = (const float*)d_in[2];
  const float* cosv  = (const float*)d_in[3];
  const float* wq    = (const float*)d_in[4];
  const float* wk    = (const float*)d_in[5];
  const float* wv    = (const float*)d_in[6];
  const float* wo    = (const float*)d_in[7];
  const float* gains = (const float*)d_in[8];
  float* outp = (float*)d_out;

  char* ws = (char*)d_ws;
  size_t off = 0;
  auto wsalloc = [&](size_t bytes) -> void* {
    void* p = (void*)(ws + off);
    off += (bytes + 255) & ~(size_t)255;
    return p;
  };
  unsigned short* xn      = (unsigned short*)wsalloc((size_t)M_ * DIM_ * 2);
  unsigned short* wq_bf   = (unsigned short*)wsalloc((size_t)DIM_ * DIM_ * 2);
  unsigned short* wk_bf   = (unsigned short*)wsalloc((size_t)HD_ * DIM_ * 2);
  unsigned short* wv_bf   = (unsigned short*)wsalloc((size_t)HD_ * DIM_ * 2);
  unsigned short* wo_bf   = (unsigned short*)wsalloc((size_t)DIM_ * DIM_ * 2);
  float*          q_f     = (float*)wsalloc((size_t)M_ * DIM_ * 4);
  float*          k_f     = (float*)wsalloc((size_t)M_ * HD_ * 4);
  float*          v_f     = (float*)wsalloc((size_t)M_ * HD_ * 4);
  unsigned short* q_rot   = (unsigned short*)wsalloc((size_t)M_ * DIM_ * 2);
  unsigned short* k_rot   = (unsigned short*)wsalloc((size_t)M_ * HD_ * 2);
  unsigned short* vT      = (unsigned short*)wsalloc((size_t)M_ * HD_ * 2);
  unsigned short* attn_bf = (unsigned short*)wsalloc((size_t)M_ * DIM_ * 2);

  rmsnorm_bf16_kernel<<<M_, 256, 0, stream>>>(x, gains, xn);
  { int n = DIM_ * DIM_; cvt_bf16_kernel<<<(n + 255) / 256, 256, 0, stream>>>(wq, wq_bf, n); }
  { int n = HD_ * DIM_;  cvt_bf16_kernel<<<(n + 255) / 256, 256, 0, stream>>>(wk, wk_bf, n); }
  { int n = HD_ * DIM_;  cvt_bf16_kernel<<<(n + 255) / 256, 256, 0, stream>>>(wv, wv_bf, n); }
  { int n = DIM_ * DIM_; cvt_bf16_kernel<<<(n + 255) / 256, 256, 0, stream>>>(wo, wo_bf, n); }

  gemm_bf16_wmma_kernel<<<dim3(DIM_ / 128, M_ / 64), 256, 0, stream>>>(
      xn, wq_bf, q_f, nullptr, nullptr, M_, DIM_, DIM_);
  gemm_bf16_wmma_kernel<<<dim3(HD_ / 128, M_ / 64), 256, 0, stream>>>(
      xn, wk_bf, k_f, nullptr, nullptr, M_, HD_, DIM_);
  gemm_bf16_wmma_kernel<<<dim3(HD_ / 128, M_ / 64), 256, 0, stream>>>(
      xn, wv_bf, v_f, nullptr, nullptr, M_, HD_, DIM_);

  rope_pack_kernel<<<dim3(S_, B_), 128, 0, stream>>>(
      q_f, k_f, v_f, sinv, cosv, q_rot, k_rot, vT);

  flash_attn_wmma_kernel<<<dim3(S_ / 16, NH_, B_), 32, 0, stream>>>(
      q_rot, k_rot, vT, attn_bf);

  gemm_bf16_wmma_kernel<<<dim3(DIM_ / 128, M_ / 64), 256, 0, stream>>>(
      attn_bf, wo_bf, outp, nullptr, x, M_, DIM_, DIM_);
}
